// Glm4MoeTopkRouter_73830487818719
// MI455X (gfx1250) — matrix operand order repaired and checked by reading, not compiled, not run on hardware
//
#include <hip/hip_runtime.h>
#include <hip/hip_bf16.h>
#include <stdint.h>

// ---------------------------------------------------------------------------
// GLM4-MoE top-k router, fused for gfx1250 (MI455X).
//  Phase 0: swizzle weight[128][4096] fp32 -> d_ws as bf16 hi/lo blocks laid
//           out per WMMA B-matrix (32x16 bf16) lane, k = 16*half + j (contiguous).
//  Phase 1: fused GEMM (bf16x3 split, v_wmma_f32_16x16x32_bf16) + routing.
//           Weight chunks staged LDS via the Tensor Data Mover (TDM),
//           double buffered, synced with s_wait_tensorcnt.
// ---------------------------------------------------------------------------

typedef __attribute__((ext_vector_type(16))) __bf16  v16bf;
typedef __attribute__((ext_vector_type(8)))  float   v8f;
typedef __attribute__((ext_vector_type(4)))  float   f32x4;
typedef __attribute__((ext_vector_type(4)))  unsigned int u32x4;
typedef __attribute__((ext_vector_type(8)))  int     i32x8;
typedef __attribute__((ext_vector_type(4)))  int     i32x4;

#define HDIM   4096
#define NEXP   128
#define NGROUP 4
#define GSIZE  32
#define TOPK   8
#define WG_TOKENS 128      // 8 waves x 16 tokens
#define KC     128         // K per stage
#define NSTAGE (HDIM / KC) // 32
#define NCHUNK (KC / 32)   // 4 k-chunks of 32 per stage
#define STAGE_BYTES 65536  // 4 chunks * 8 tiles * 2(hi/lo) * 32 lanes * 32B

// LDS map (dynamic shared):
//   [0, 512)          bias (pad to 1024)
//   [1024, +2*64KB)   weight double buffer
//   [132096, +64KB)   logits 128 tokens x 128 experts fp32
#define LDS_WBUF   1024
#define LDS_LOGITS (1024 + 2 * STAGE_BYTES)
#define LDS_TOTAL  (LDS_LOGITS + WG_TOKENS * NEXP * 4)

// ---------------------------------------------------------------------------
// Phase 0: convert + swizzle weight into d_ws.
// Layout: per (chunk c of 32 k, tile t of 16 experts):
//   [hi: 32 lanes x 32B][lo: 32 lanes x 32B]  (2 KB per (c,t))
// lane L, half j holds W[e = t*16 + (L%16)][k = c*32 + (L/16)*16 + j]
// contiguous k order (not AMD's table order); A uses the same order.
// ---------------------------------------------------------------------------
__global__ void swizzle_weight_kernel(const float* __restrict__ w,
                                      unsigned char* __restrict__ ws) {
  int u    = blockIdx.x * blockDim.x + threadIdx.x;   // 0 .. 128*8*32-1
  int chunk = u >> 8;
  int rem   = u & 255;
  int tile  = rem >> 5;
  int lane  = rem & 31;
  int e = tile * 16 + (lane & 15);
  int k = chunk * 32 + (lane >> 4) * 16;
  const float* src = w + (size_t)e * HDIM + k;
  v16bf hv, lv;
#pragma unroll
  for (int j = 0; j < 16; ++j) {
    float x = src[j];
    __bf16 h = (__bf16)x;
    hv[j] = h;
    lv[j] = (__bf16)(x - (float)h);
  }
  size_t base = ((size_t)(chunk * 8 + tile) * 2) * 1024 + (size_t)lane * 32;
  *reinterpret_cast<v16bf*>(ws + base)        = hv;
  *reinterpret_cast<v16bf*>(ws + base + 1024) = lv;
}

// ---------------------------------------------------------------------------
// TDM: linear 64 KB copy global -> LDS (one descriptor, data_size=8B,
// tensor_dim0 = tile_dim0 = 8192 elements, 1D tile; higher dims unused=0).
// Issued by wave 0 only; completion via s_wait_tensorcnt + barrier.
// ---------------------------------------------------------------------------
__device__ __forceinline__ void tdm_copy_stage(uint64_t gaddr, unsigned int lds_off) {
  u32x4 g0 = { 1u,                                    // count=1 (user D#)
               lds_off,                               // lds_addr (bytes)
               (unsigned int)(gaddr & 0xffffffffu),   // global_addr[31:0]
               (unsigned int)((gaddr >> 32) & 0x1ffffffu) | 0x80000000u }; // [56:32] | type=2
  i32x8 g1 = { 0x00030000,   // wg_mask=0, data_size=3 (8B), no flags
               0x20000000,   // tensor_dim0[15:0]=8192 -> bits[79:64->no: 63:48] of group
               0,            // tensor_dim0 hi / tensor_dim1 lo
               0x20000000,   // tile_dim0 = 8192 (bits 127:112)
               0,            // tile_dim1 = tile_dim2 = 0 (unused -> 1D)
               0x00002000,   // tensor_dim0_stride = 8192
               0, 0 };
  i32x4 gz = { 0, 0, 0, 0 };
#if __has_include(<hip/amd_detail/amd_gfx1250_TDM.h>)
  i32x8 gz8 = { 0, 0, 0, 0, 0, 0, 0, 0 };
  __builtin_amdgcn_tensor_load_to_lds(g0, g1, gz, gz, gz8, 0);
#else
  __builtin_amdgcn_tensor_load_to_lds(g0, g1, gz, gz, 0);
#endif
}

// ---------------------------------------------------------------------------
// Phase 1: fused router. One workgroup = 256 threads = 8 waves = 128 tokens.
// ---------------------------------------------------------------------------
__global__ void
__launch_bounds__(256)
glm4_router_kernel(const float* __restrict__ hs,
                   const float* __restrict__ bias,
                   const unsigned char* __restrict__ wswz,
                   int* __restrict__ idx_out,
                   float* __restrict__ w_out,
                   int T) {
  extern __shared__ __align__(128) unsigned char smem[];
  float*         sBias   = reinterpret_cast<float*>(smem);
  unsigned char* sW      = smem + LDS_WBUF;
  float*         sLogit  = reinterpret_cast<float*>(smem + LDS_LOGITS);

  const int tid  = threadIdx.x;
  const int wave = tid >> 5;
  const int lane = tid & 31;
  const int m    = lane & 15;
  const int half = lane >> 4;
  const int tokBlock = blockIdx.x * WG_TOKENS;

  if (tid < NEXP) sBias[tid] = bias[tid];

  // LDS byte offsets of the two weight buffers (dynamic LDS base is 0).
  const unsigned int ldsW0 = (unsigned int)(size_t)(sW);
  const unsigned int ldsW1 = (unsigned int)(size_t)(sW + STAGE_BYTES);

  // Prefetch stage 0 via TDM (wave 0).
  if (wave == 0) {
    tdm_copy_stage((uint64_t)(uintptr_t)wswz, ldsW0);
  }

  v8f acc[8];
#pragma unroll
  for (int t = 0; t < 8; ++t) acc[t] = (v8f)(0.0f);

  const float* rowPtr = hs + (size_t)(tokBlock + wave * 16 + m) * HDIM;

  for (int s = 0; s < NSTAGE; ++s) {
    if (wave == 0) __builtin_amdgcn_s_wait_tensorcnt(0);
    __syncthreads();
    // Prefetch next stage into the other buffer (its readers finished at the
    // barrier above).
    if (s + 1 < NSTAGE && wave == 0) {
      tdm_copy_stage((uint64_t)(uintptr_t)(wswz + (size_t)(s + 1) * STAGE_BYTES),
                     ((s + 1) & 1) ? ldsW1 : ldsW0);
    }
    const unsigned char* wb = sW + (size_t)(s & 1) * STAGE_BYTES;

    for (int c = 0; c < NCHUNK; ++c) {
      const int k = s * KC + c * 32;
      // A (16x32 bf16) lane order used here, same as B in ws: lane m (half 0): K 0..15 ;
      // lane m+16 (half 1): K 16..31.
      const f32x4* p0 = reinterpret_cast<const f32x4*>(rowPtr + k + half * 16);
      const f32x4* p1 = reinterpret_cast<const f32x4*>(rowPtr + k + half * 16 + 8);
      f32x4 a0 = __builtin_nontemporal_load(p0);
      f32x4 a1 = __builtin_nontemporal_load(p0 + 1);
      f32x4 b0 = __builtin_nontemporal_load(p1);
      f32x4 b1 = __builtin_nontemporal_load(p1 + 1);
      float xv[16];
#pragma unroll
      for (int j = 0; j < 4; ++j) {
        xv[j]      = a0[j];
        xv[4 + j]  = a1[j];
        xv[8 + j]  = b0[j];
        xv[12 + j] = b1[j];
      }
      v16bf Ahi, Alo;
#pragma unroll
      for (int j = 0; j < 16; ++j) {
        __bf16 h = (__bf16)xv[j];
        Ahi[j] = h;
        Alo[j] = (__bf16)(xv[j] - (float)h);
      }
#pragma unroll
      for (int t = 0; t < 8; ++t) {
        const unsigned char* cb = wb + (size_t)((c * 8 + t) * 2) * 1024 + (size_t)lane * 32;
        v16bf Bh = *reinterpret_cast<const v16bf*>(cb);
        v16bf Bl = *reinterpret_cast<const v16bf*>(cb + 1024);
        // fp32 via bf16 split: hi*hi + hi*lo + lo*hi
        acc[t] = __builtin_amdgcn_wmma_f32_16x16x32_bf16(
            false, Ahi, false, Bh, (short)0, acc[t], false, false);
        acc[t] = __builtin_amdgcn_wmma_f32_16x16x32_bf16(
            false, Ahi, false, Bl, (short)0, acc[t], false, false);
        acc[t] = __builtin_amdgcn_wmma_f32_16x16x32_bf16(
            false, Alo, false, Bh, (short)0, acc[t], false, false);
      }
    }
  }

  // Scatter logits to LDS. D layout: VGPR r -> token row (r + 8*half),
  // lane%16 -> expert column within tile.
#pragma unroll
  for (int t = 0; t < 8; ++t) {
#pragma unroll
    for (int r = 0; r < 8; ++r) {
      int lt = wave * 16 + r + 8 * half;
      int e  = t * 16 + m;
      sLogit[lt * NEXP + e] = acc[t][r];
    }
  }
  __syncthreads();

  // ---- Routing epilogue: one thread per token (threads 0..127). ----
  if (tid < WG_TOKENS) {
    float* Lr = sLogit + tid * NEXP;
    // sigmoid in place (raw scores; score_for_choice = score + bias on the fly)
    for (int e = 0; e < NEXP; ++e) {
      Lr[e] = 1.0f / (1.0f + __expf(-Lr[e]));
    }
    // group score = sum of top-2 (score+bias) within each group of 32
    float gs[NGROUP];
#pragma unroll
    for (int g = 0; g < NGROUP; ++g) {
      float m1 = -3.0e38f, m2 = -3.0e38f;
      for (int j = 0; j < GSIZE; ++j) {
        int e = g * GSIZE + j;
        float v = Lr[e] + sBias[e];
        if (v > m1) { m2 = m1; m1 = v; }
        else if (v > m2) { m2 = v; }
      }
      gs[g] = m1 + m2;
    }
    // top-2 groups (first-index wins ties, matching lax.top_k)
    int g0 = 0;
    {
      float b = gs[0];
      for (int g = 1; g < NGROUP; ++g) if (gs[g] > b) { b = gs[g]; g0 = g; }
    }
    int g1 = -1;
    {
      float b = -3.0e38f;
      for (int g = 0; g < NGROUP; ++g) {
        if (g == g0) continue;
        if (gs[g] > b) { b = gs[g]; g1 = g; }
      }
    }
    const unsigned allowed = (1u << g0) | (1u << g1);
    // top-8 over masked scores (masked-out experts contribute 0.0, as in ref)
    unsigned pm[4] = {0u, 0u, 0u, 0u};
    int   idxs[TOPK];
    float wv[TOPK];
    float wsum = 0.0f;
    for (int i = 0; i < TOPK; ++i) {
      float best = -3.0e38f;
      int   bi   = 0;
      for (int e = 0; e < NEXP; ++e) {
        if ((pm[e >> 5] >> (e & 31)) & 1u) continue;
        float v = ((allowed >> (e >> 5)) & 1u) ? (Lr[e] + sBias[e]) : 0.0f;
        if (v > best) { best = v; bi = e; }
      }
      pm[bi >> 5] |= 1u << (bi & 31);
      idxs[i] = bi;
      float sw = Lr[bi];   // gather from raw sigmoid scores
      wv[i] = sw;
      wsum += sw;
    }
    const float inv = 1.0f / (wsum + 1e-20f);  // ROUTED_SCALING_FACTOR == 1.0
    const size_t gt = (size_t)(tokBlock + tid);
#pragma unroll
    for (int i = 0; i < TOPK; ++i) {
      idx_out[gt * TOPK + i] = idxs[i];
      w_out[gt * TOPK + i]   = wv[i] * inv;
    }
  }
}

// ---------------------------------------------------------------------------
extern "C" void kernel_launch(void* const* d_in, const int* in_sizes, int n_in,
                              void* d_out, int out_size, void* d_ws, size_t ws_size,
                              hipStream_t stream) {
  const float* hs   = (const float*)d_in[0];   // [T, 4096] fp32
  const float* wgt  = (const float*)d_in[1];   // [128, 4096] fp32
  const float* bias = (const float*)d_in[2];   // [128] fp32
  const int T = in_sizes[0] / HDIM;

  unsigned char* ws = (unsigned char*)d_ws;    // needs 2 MB (swizzled weight)

  // Phase 0: swizzle weight (128 chunks * 8 tiles * 32 lanes = 32768 threads)
  swizzle_weight_kernel<<<128, 256, 0, stream>>>(wgt, ws);

  // Outputs: [T,8] int32 indices then [T,8] fp32 weights, concatenated.
  int*   idx_out = reinterpret_cast<int*>(d_out);
  float* w_out   = reinterpret_cast<float*>(d_out) + (size_t)T * TOPK;

  glm4_router_kernel<<<T / WG_TOKENS, 256, LDS_TOTAL, stream>>>(
      hs, bias, ws, idx_out, w_out, T);
}
